// Flow1_74895639707748
// MI455X (gfx1250) — matrix operand order repaired and checked by reading, not compiled, not run on hardware
//
#include <hip/hip_runtime.h>
#include <hip/hip_bf16.h>
#include <math.h>

typedef __attribute__((ext_vector_type(16))) _Float16 v16h;
typedef __attribute__((ext_vector_type(8)))  float    v8f;

#define KS      32
#define BATCH   4096
#define ZSDIM   128
#define ZH      64
#define HS      50
#define NFLOW   2
#define NROWS   (KS * BATCH)          // 131072
#define WAVES   4                     // waves per block
#define ROWS_PB (WAVES * 16)          // rows per block = 64
#define ZSTRIDE 132                   // padded f32 row stride (bank-conflict avoidance)

__device__ inline void lds_fence() {
    asm volatile("s_wait_dscnt 0" ::: "memory");
}

__device__ inline v8f wmma_f16(v16h a, v16h b, v8f c) {
    return __builtin_amdgcn_wmma_f32_16x16x32_f16(
        /*neg_a=*/false, a, /*neg_b=*/false, b,
        /*c_mod=*/(short)0, c, /*reuse_a=*/false, /*reuse_b=*/false);
}

// branch-free tanh via native v_exp_f32 / v_rcp_f32 (saturates correctly at +/-1)
__device__ inline float tanh_fast(float x) {
    float e = __expf(x + x);
    return 1.f - 2.f * __builtin_amdgcn_rcpf(e + 1.f);
}

__launch_bounds__(WAVES * 32)
__global__ void flow_fused_kernel(const float* __restrict__ mean,
                                  const float* __restrict__ logvar,
                                  const float* __restrict__ eps,
                                  const float* __restrict__ W0,
                                  const float* __restrict__ b0,
                                  const float* __restrict__ W1,
                                  const float* __restrict__ b1,
                                  const float* __restrict__ W2,
                                  const float* __restrict__ b2,
                                  float* __restrict__ z_out,
                                  float* __restrict__ logpz) {
    __shared__ __align__(16) float     zbuf[WAVES][16][ZSTRIDE]; // fp32 tile state z1||z2
    __shared__ __align__(16) _Float16  hbuf[WAVES][16][64];      // f16 hidden activations
    __shared__ __align__(16) float     part[WAVES][8][32];       // logdet partials

    const int lane = threadIdx.x & 31;
    const int wv   = threadIdx.x >> 5;
    const int rl   = lane & 15;        // local row (phase 1/3)
    const int hi   = lane >> 4;        // which 64-col half this lane streams (phase 1/3)

    float (*zt)[ZSTRIDE] = zbuf[wv];
    _Float16 (*ht)[64]   = hbuf[wv];
    float (*pt)[32]      = part[wv];

    const int grow = blockIdx.x * ROWS_PB + wv * 16 + rl;   // global row, exact fit
    const int brow = grow & (BATCH - 1);                    // broadcast row for mean/logvar

    // ---------------- Phase 1: z = eps*exp(0.5*logvar)+mean ; Gaussian logq ----------
    const float* epsr  = eps    + (size_t)grow * ZSDIM + hi * 64;
    const float* meanr = mean   + (size_t)brow * ZSDIM + hi * 64;
    const float* lvr   = logvar + (size_t)brow * ZSDIM + hi * 64;
    float s = 0.f;
#pragma unroll
    for (int j = 0; j < 16; ++j) {
        float4 e4 = ((const float4*)epsr)[j];
        float4 m4 = ((const float4*)meanr)[j];
        float4 l4 = ((const float4*)lvr)[j];
        float4 z4;
        z4.x = e4.x * __expf(0.5f * l4.x) + m4.x;  s += l4.x + e4.x * e4.x;
        z4.y = e4.y * __expf(0.5f * l4.y) + m4.y;  s += l4.y + e4.y * e4.y;
        z4.z = e4.z * __expf(0.5f * l4.z) + m4.z;  s += l4.z + e4.z * e4.z;
        z4.w = e4.w * __expf(0.5f * l4.w) + m4.w;  s += l4.w + e4.w * e4.w;
        ((float4*)&zt[rl][hi * 64])[j] = z4;
    }
    s += __shfl_xor(s, 16, 32);
    const float logq = -0.5f * ((float)ZSDIM * 1.8378770664093453f + s);
    float logdet = 0.f;

    // -------------- Phase 2: NF coupling flows, each with two half-updates -----------
    const int m    = lane & 15;     // fragment row (A) / fragment column (B, C/D)
    const int kreg = lane >> 4;     // K-region select: k = 16*kreg + e within each 32-k A/B frag (contiguous order)

    for (int fi = 0; fi < NFLOW; ++fi) {
        for (int hf = 0; hf < 2; ++hf) {
            const int srcoff = hf ? 64 : 0;
            const int dstoff = hf ? 0 : 64;
            const int wsel   = fi * 2 + hf;
            const float* W0p = W0 + (size_t)wsel * HS * ZH;
            const float* b0p = b0 + (size_t)wsel * HS;
            const float* W1p = W1 + (size_t)wsel * ZH * HS;
            const float* b1p = b1 + (size_t)wsel * ZH;
            const float* W2p = W2 + (size_t)wsel * ZH * HS;
            const float* b2p = b2 + (size_t)wsel * ZH;

            lds_fence();
            // ---- A fragments of src half (16x64 f32 -> two 16x32 f16 frags) ----
            v16h a0, a1;
            {
                const float* p = &zt[m][srcoff + kreg * 16];
#pragma unroll
                for (int e = 0; e < 8; ++e) {
                    a0[e]     = (_Float16)p[e];
                    a0[8 + e] = (_Float16)p[8 + e];
                    a1[e]     = (_Float16)p[32 + e];
                    a1[8 + e] = (_Float16)p[40 + e];
                }
            }

            // ---- Step A: h = tanh(src @ W0^T + b0), stored f16 (cols >= HS zeroed) ----
#pragma unroll
            for (int t = 0; t < 4; ++t) {
                const int  n  = t * 16 + m;        // B column / C column
                const bool nv = (n < HS);
                v16h bf0, bf1;
                {
                    const float* wp = W0p + n * ZH + kreg * 16;
#pragma unroll
                    for (int e = 0; e < 16; ++e) {
                        bf0[e] = nv ? (_Float16)wp[e]      : (_Float16)0.f;
                        bf1[e] = nv ? (_Float16)wp[32 + e] : (_Float16)0.f;
                    }
                }
                const float bias = nv ? b0p[n] : 0.f;
                v8f acc;
#pragma unroll
                for (int r = 0; r < 8; ++r) acc[r] = bias;
                acc = wmma_f16(a0, bf0, acc);
                acc = wmma_f16(a1, bf1, acc);
#pragma unroll
                for (int r = 0; r < 8; ++r) {
                    const float hv = nv ? tanh_fast(acc[r]) : 0.f;
                    ht[r + 8 * kreg][t * 16 + m] = (_Float16)hv;
                }
            }

            lds_fence();
            // ---- A fragments of h (16x64 f16, cols >= HS are zero) ----
            v16h ha0, ha1;
            {
                const _Float16* p = &ht[m][kreg * 16];
#pragma unroll
                for (int e = 0; e < 8; ++e) {
                    ha0[e]     = p[e];
                    ha0[8 + e] = p[8 + e];
                    ha1[e]     = p[32 + e];
                    ha1[8 + e] = p[40 + e];
                }
            }

            // ---- Step B: mew/sig GEMMs + affine update of dst half + logdet ----
            float pl[8];
#pragma unroll
            for (int r = 0; r < 8; ++r) pl[r] = 0.f;

#pragma unroll
            for (int t = 0; t < 4; ++t) {
                const int n = t * 16 + m;          // dst column 0..63, always valid
                v16h w1f0, w1f1, w2f0, w2f1;
                {
                    const float* p1  = W1p + n * HS;
                    const float* p2  = W2p + n * HS;
                    const int    kb0 = kreg * 16;
                    const int    kb1 = 32 + kreg * 16;
#pragma unroll
                    for (int e = 0; e < 16; ++e) {
                        const int k0 = kb0 + e, k1 = kb1 + e;
                        w1f0[e] = (k0 < HS) ? (_Float16)p1[k0] : (_Float16)0.f;
                        w1f1[e] = (k1 < HS) ? (_Float16)p1[k1] : (_Float16)0.f;
                        w2f0[e] = (k0 < HS) ? (_Float16)p2[k0] : (_Float16)0.f;
                        w2f1[e] = (k1 < HS) ? (_Float16)p2[k1] : (_Float16)0.f;
                    }
                }
                v8f ma, sa;
                const float bm = b1p[n], bs = b2p[n];
#pragma unroll
                for (int r = 0; r < 8; ++r) { ma[r] = bm; sa[r] = bs; }
                ma = wmma_f16(ha0, w1f0, ma);
                ma = wmma_f16(ha1, w1f1, ma);
                sa = wmma_f16(ha0, w2f0, sa);
                sa = wmma_f16(ha1, w2f1, sa);
#pragma unroll
                for (int r = 0; r < 8; ++r) {
                    // sigma = 1/(1+exp(-x)); log(sigma) = -log(1+exp(-x))
                    const float e  = __expf(-sa[r]);
                    const float tden = 1.f + e;
                    const float sg = __builtin_amdgcn_rcpf(tden);
                    float* zp = &zt[r + 8 * kreg][dstoff + n];
                    const float zv = *zp;
                    *zp = zv * sg + ma[r];
                    pl[r] -= __logf(tden);
                }
            }

            // ---- logdet reduction across the 16 fragment columns ----
#pragma unroll
            for (int r = 0; r < 8; ++r) pt[r][lane] = pl[r];
            lds_fence();
            if (lane < 16) {
                const int rr = lane & 7;
                const int bo = (lane >> 3) * 16;
                float rs = 0.f;
#pragma unroll
                for (int j = 0; j < 16; ++j) rs += pt[rr][bo + j];
                logdet += rs;
            }
        }
    }

    // ---------------- Phase 3: write z_out and logpz ----------------------------------
    lds_fence();
    float* zo = z_out + (size_t)grow * ZSDIM + hi * 64;
#pragma unroll
    for (int j = 0; j < 16; ++j)
        ((float4*)zo)[j] = ((const float4*)&zt[rl][hi * 64])[j];
    if (lane < 16)
        logpz[grow] = logq - logdet;
}

extern "C" void kernel_launch(void* const* d_in, const int* in_sizes, int n_in,
                              void* d_out, int out_size, void* d_ws, size_t ws_size,
                              hipStream_t stream) {
    const float* mean   = (const float*)d_in[0];
    const float* logvar = (const float*)d_in[1];
    const float* eps    = (const float*)d_in[2];
    const float* W0     = (const float*)d_in[3];
    const float* b0     = (const float*)d_in[4];
    const float* W1     = (const float*)d_in[5];
    const float* b1     = (const float*)d_in[6];
    const float* W2     = (const float*)d_in[7];
    const float* b2     = (const float*)d_in[8];
    float* z_out = (float*)d_out;
    float* logpz = z_out + (size_t)NROWS * ZSDIM;

    const int blocks = NROWS / ROWS_PB;   // 2048
    flow_fused_kernel<<<blocks, WAVES * 32, 0, stream>>>(
        mean, logvar, eps, W0, b0, W1, b1, W2, b2, z_out, logpz);
}
